// LayerNormDenseGeneral_1563368095882
// MI455X (gfx1250) — matrix operand order repaired and checked by reading, not compiled, not run on hardware
//
#include <hip/hip_runtime.h>

// ---------------------------------------------------------------------------
// LayerNorm (H=1024) + DenseGeneral GEMM (32768x1024 @ 1024x1024) for gfx1250.
//   - GEMM uses v_wmma_f32_16x16x32_bf16 (bf16 operands, f32 accumulate).
//   - B tiles staged into LDS by the Tensor Data Mover (tensor_load_to_lds),
//     double-buffered, waited with s_wait_tensorcnt.
//   - A pre-converted to bf16 by the LN kernel; A loads software-pipelined.
//   - All 8 B fragments preloaded per K-step; K-loop kept at unroll 1 to
//     stay under the VGPR budget (no accumulator spills).
// d_out = [ out (32768*1024 f32) | ln_out (32768*1024 f32) ]
// d_ws  = [ A_bf16 (64 MB) | W_bf16 (2 MB) ]
// ---------------------------------------------------------------------------

typedef __attribute__((ext_vector_type(16))) __bf16          v16bf;
typedef __attribute__((ext_vector_type(8)))  float           v8f;
typedef __attribute__((ext_vector_type(16))) unsigned short  v16u;
typedef __attribute__((ext_vector_type(8)))  unsigned short  v8u;
typedef __attribute__((ext_vector_type(4)))  unsigned short  v4u;
typedef __attribute__((ext_vector_type(4)))  unsigned int    u32x4;
typedef __attribute__((ext_vector_type(8)))  int             i32x8;
typedef __attribute__((ext_vector_type(4)))  int             i32x4;

#define HDIM  1024
#define FDIM  1024
#define NROWS 32768           // S*B = 4096*8
#define LN_EPS 1e-6f

// LDS row stride for the B tile: 256 data bytes + 16 pad bytes
// (TDM pad: 4 dwords every 64 dwords) -> keeps 16B ds alignment,
// spreads banks (2-way worst case instead of 32-way).
#define BROW_BYTES 272

// float -> bf16 bits, round-to-nearest-even
__device__ __forceinline__ unsigned short f2bf(float f) {
  union { float f; unsigned u; } v; v.f = f;
  unsigned u = v.u;
  unsigned r = (u + 0x7FFFu + ((u >> 16) & 1u)) >> 16;
  return (unsigned short)r;
}

// ---------------------------------------------------------------------------
// Kernel 1: convert f32 weights (H x F row-major) to bf16 bits.
// ---------------------------------------------------------------------------
__global__ void wconv_kernel(const float* __restrict__ w,
                             unsigned short* __restrict__ wb) {
  int i = (blockIdx.x * blockDim.x + threadIdx.x) * 8;
  float4 a = *(const float4*)(w + i);
  float4 b = *(const float4*)(w + i + 4);
  v8u o;
  o[0] = f2bf(a.x); o[1] = f2bf(a.y); o[2] = f2bf(a.z); o[3] = f2bf(a.w);
  o[4] = f2bf(b.x); o[5] = f2bf(b.y); o[6] = f2bf(b.z); o[7] = f2bf(b.w);
  *(v8u*)(wb + i) = o;
}

// ---------------------------------------------------------------------------
// Kernel 2: LayerNorm, one row per 256-thread block. Writes f32 ln_out
// (second returned tensor) and a packed bf16 copy (GEMM A operand).
// ---------------------------------------------------------------------------
__global__ void ln_kernel(const float* __restrict__ x,
                          const float* __restrict__ gamma,
                          const float* __restrict__ beta,
                          float* __restrict__ lnout,
                          unsigned short* __restrict__ abf) {
  __shared__ float sm[8];
  const int row = blockIdx.x;
  const int t   = threadIdx.x;
  const float* xr = x + (size_t)row * HDIM;

  float4 xv = *(const float4*)(xr + t * 4);

  // ---- mean ----
  float s = xv.x + xv.y + xv.z + xv.w;
  #pragma unroll
  for (int o = 16; o > 0; o >>= 1) s += __shfl_xor(s, o, 32);
  if ((t & 31) == 0) sm[t >> 5] = s;
  __syncthreads();
  float tot = 0.0f;
  #pragma unroll
  for (int i = 0; i < 8; ++i) tot += sm[i];
  const float mu = tot * (1.0f / HDIM);

  // ---- variance ----
  float d0 = xv.x - mu, d1 = xv.y - mu, d2 = xv.z - mu, d3 = xv.w - mu;
  float ss = d0 * d0 + d1 * d1 + d2 * d2 + d3 * d3;
  __syncthreads();
  #pragma unroll
  for (int o = 16; o > 0; o >>= 1) ss += __shfl_xor(ss, o, 32);
  if ((t & 31) == 0) sm[t >> 5] = ss;
  __syncthreads();
  float vtot = 0.0f;
  #pragma unroll
  for (int i = 0; i < 8; ++i) vtot += sm[i];
  const float rstd = rsqrtf(vtot * (1.0f / HDIM) + LN_EPS);

  float4 g  = *(const float4*)(gamma + t * 4);
  float4 bb = *(const float4*)(beta  + t * 4);
  float4 y;
  y.x = d0 * rstd * g.x + bb.x;
  y.y = d1 * rstd * g.y + bb.y;
  y.z = d2 * rstd * g.z + bb.z;
  y.w = d3 * rstd * g.w + bb.w;
  *(float4*)(lnout + (size_t)row * HDIM + t * 4) = y;

  v4u yb;
  yb[0] = f2bf(y.x); yb[1] = f2bf(y.y); yb[2] = f2bf(y.z); yb[3] = f2bf(y.w);
  *(v4u*)(abf + (size_t)row * HDIM + t * 4) = yb;
}

// ---------------------------------------------------------------------------
// Kernel 3: WMMA GEMM, block tile 128(M) x 128(N), 8 waves.
// B slab (32K x 128N bf16) staged to LDS per K-step by the TDM,
// double-buffered; fragments read via ds_load; A software-pipelined.
// ---------------------------------------------------------------------------
__global__ void gemm_kernel(const unsigned short* __restrict__ abf,
                            const unsigned short* __restrict__ wbf,
                            float* __restrict__ out) {
  __shared__ __align__(16) unsigned char bbuf[2][32 * BROW_BYTES];

  const int lane = threadIdx.x & 31;
  const int wv   = threadIdx.x >> 5;
  const int n0   = blockIdx.x * 128;
  const int m0   = blockIdx.y * 128;

  // A-fragment addressing (16-bit A 16x32 layout, ISA 7.12.2):
  // lanes 0-15 -> M=lane, K {0..7,16..23}; lanes 16-31 -> M=lane-16, K {8..15,24..31}
  const int mrow = m0 + wv * 16 + (lane & 15);
  const int hiK  = (lane >> 4) * 8;
  const unsigned short* arow = abf + (size_t)mrow * HDIM;

  v8f acc[8] = {};

  // ---- TDM descriptor issue: load W rows [k0,k0+32) cols [n0,n0+128) ----
  auto issue_tdm = [&](int k0, int buf) {
    unsigned long long gaddr =
        (unsigned long long)(const void*)(wbf + (size_t)k0 * FDIM + n0);
    unsigned int lds_off =
        (unsigned int)(unsigned long long)(void*)&bbuf[buf][0];  // addr[31:0] = LDS byte offset

    u32x4 g0;
    g0[0] = 1u;                                         // count=1, user descriptor
    g0[1] = lds_off;                                    // lds_addr
    g0[2] = (unsigned int)gaddr;                        // global_addr[31:0]
    g0[3] = (unsigned int)(gaddr >> 32) | (2u << 30);   // global_addr[56:32] | type=2

    i32x8 g1;
    g1[0] = (1 << 16)        // data_size = 2 bytes
          | (1 << 20)        // pad_enable
          | (5 << 22)        // pad_interval: every 64 dwords (=256B row)
          | (3 << 25);       // pad_amount: 4 dwords (16B) -> 272B row stride
    g1[1] = (FDIM & 0xFFFF) << 16;                        // tensor_dim0[15:0]
    g1[2] = ((FDIM >> 16) & 0xFFFF)
          | ((HDIM & 0xFFFF) << 16);                      // dim0[31:16] | dim1[15:0]
    g1[3] = ((HDIM >> 16) & 0xFFFF) | (128 << 16);        // dim1[31:16] | tile_dim0=128
    g1[4] = 32;                                           // tile_dim1=32, tile_dim2=0
    g1[5] = FDIM;                                         // tensor_dim0_stride[31:0]
    g1[6] = 0;                                            // stride0 hi, stride1 lo (unused, 2D)
    g1[7] = 0;
    // 6-arg toolchain variant: (g0, g1, g2, g3, g4, cpol)
    __builtin_amdgcn_tensor_load_to_lds(g0, g1, (i32x4)0, (i32x4)0, (i32x8)0, 0);
  };

  if (wv == 0) issue_tdm(0, 0);

  // A software pipeline: registers hold the A fragment for the CURRENT step.
  v8u a0 = *(const v8u*)(arow + hiK);
  v8u a1 = *(const v8u*)(arow + hiK + 16);

  const int nsteps = HDIM / 32;                 // 32 K-steps
  #pragma unroll 1                              // keep VGPRs < 256: no spills
  for (int i = 0; i < nsteps; ++i) {
    const int k0 = i * 32;

    if (wv == 0) {
      if (i + 1 < nsteps) {
        issue_tdm(k0 + 32, (i + 1) & 1);        // prefetch next slab via TDM
        __builtin_amdgcn_s_wait_tensorcnt(1);   // current slab done, next in flight
      } else {
        __builtin_amdgcn_s_wait_tensorcnt(0);
      }
    }
    __syncthreads();                            // publish LDS slab to all waves

    // Assemble current A fragment, then issue next step's A loads so their
    // latency hides behind this step's WMMAs.
    v16u au = __builtin_shufflevector(a0, a1,
                0, 1, 2, 3, 4, 5, 6, 7, 8, 9, 10, 11, 12, 13, 14, 15);
    v16bf afrag = __builtin_bit_cast(v16bf, au);
    if (i + 1 < nsteps) {
      a0 = *(const v8u*)(arow + k0 + 32 + hiK);
      a1 = *(const v8u*)(arow + k0 + 32 + hiK + 16);
      __builtin_prefetch(arow + k0 + 64, 0, 0); // global_prefetch_b8 for step i+2
    }

    // Preload ALL B fragments (per-element k gathers: col n = lane&15, K {hiK..hiK+7, 16+hiK..23+hiK}), then run WMMAs back-to-back.
    const unsigned char* bb = &bbuf[i & 1][0] + (lane & 15) * 2;
    v16bf bfrag[8];
    #pragma unroll
    for (int t = 0; t < 8; ++t)
      #pragma unroll
      for (int s = 0; s < 16; ++s)
        bfrag[t][s] = *(const __bf16*)(bb + (size_t)((s < 8 ? hiK + s : 16 + hiK + (s - 8)) * BROW_BYTES) + t * 32);

    #pragma unroll
    for (int t = 0; t < 8; ++t)
      acc[t] = __builtin_amdgcn_wmma_f32_16x16x32_bf16(
          /*neg_a=*/false, afrag, /*neg_b=*/false, bfrag[t],
          /*c_mod=*/(short)0, acc[t], /*reuse_a=*/false, /*reuse_b=*/false);

    __syncthreads();                            // slab consumed; TDM may overwrite
  }

  // D layout: VGPR r -> row r (lanes 0-15) / r+8 (lanes 16-31); lane&15 = col
  const int mbase = m0 + wv * 16 + ((lane >> 4) << 3);
  const int ncol  = lane & 15;
  #pragma unroll
  for (int t = 0; t < 8; ++t) {
    float* op = out + (size_t)mbase * FDIM + n0 + t * 16 + ncol;
    #pragma unroll
    for (int r = 0; r < 8; ++r) op[(size_t)r * FDIM] = acc[t][r];
  }
}

// ---------------------------------------------------------------------------
extern "C" void kernel_launch(void* const* d_in, const int* in_sizes, int n_in,
                              void* d_out, int out_size, void* d_ws, size_t ws_size,
                              hipStream_t stream) {
  const float* x     = (const float*)d_in[0];   // (4096, 8, 1024) f32
  const float* scale = (const float*)d_in[1];   // (1024,) f32
  const float* bias  = (const float*)d_in[2];   // (1024,) f32
  const float* wker  = (const float*)d_in[3];   // (1024, 1024) f32

  float* out   = (float*)d_out;                         // first output
  float* lnout = out + (size_t)NROWS * FDIM;            // second output (ln_out)

  unsigned short* abf = (unsigned short*)d_ws;          // bf16 ln_out (64 MB)
  unsigned short* wbf = abf + (size_t)NROWS * HDIM;     // bf16 weights (2 MB)

  // 1) weights f32 -> bf16
  wconv_kernel<<<(HDIM * FDIM) / (256 * 8), 256, 0, stream>>>(wker, wbf);

  // 2) LayerNorm -> f32 ln_out + bf16 A operand
  ln_kernel<<<NROWS, 256, 0, stream>>>(x, scale, bias, lnout, abf);

  // 3) WMMA GEMM with TDM-staged B tiles
  dim3 grid(FDIM / 128, NROWS / 128);
  gemm_kernel<<<grid, 256, 0, stream>>>(abf, wbf, out);
}
